// VariationOfInformation_18820546691883
// MI455X (gfx1250) — 8-bit matrix operand order repaired and checked by reading, compiled, not run on hardware
//
#include <hip/hip_runtime.h>
#include <hip/hip_bf16.h>
#include <stdint.h>


#define NBINS   20
#define NASSET  32
#define S_TOTAL 65536
#define ROWS    (NASSET * NBINS)
#define EPS_F   1e-10f
#define SPLITK  16
#define KCHUNK  (S_TOTAL / SPLITK)

typedef __attribute__((ext_vector_type(8))) int      v8i;
typedef __attribute__((ext_vector_type(4))) uint32_t v4u;

#if __has_builtin(__builtin_amdgcn_global_load_async_to_lds_b128) && \
    __has_builtin(__builtin_amdgcn_s_wait_asynccnt)
#define HAVE_ASYNC_LDS 1
typedef int b128_t __attribute__((vector_size(16)));
typedef __attribute__((address_space(1))) b128_t* gbl_b128_p;
typedef __attribute__((address_space(3))) b128_t* lds_b128_p;
#else
#define HAVE_ASYNC_LDS 0
#endif

static const size_t WS_MINMAX = 0;
static const size_t WS_IDX8   = 256;
static const size_t WS_A      = WS_IDX8 + (size_t)NASSET * S_TOTAL;
static const size_t WS_JOINT  = WS_A + (size_t)ROWS * S_TOTAL;

__global__ void k_minmax(const float* __restrict__ x, float* __restrict__ mm) {
  __shared__ float smin[256], smax[256];
  const int n = blockIdx.x, tid = threadIdx.x;
  float mn = 3.402823466e38f, mx = -3.402823466e38f;
  for (int s = tid; s < S_TOTAL; s += 256) {
    float v = x[(size_t)s * NASSET + n];
    mn = fminf(mn, v);  mx = fmaxf(mx, v);
  }
  smin[tid] = mn; smax[tid] = mx;
  __syncthreads();
  for (int off = 128; off > 0; off >>= 1) {
    if (tid < off) {
      smin[tid] = fminf(smin[tid], smin[tid + off]);
      smax[tid] = fmaxf(smax[tid], smax[tid + off]);
    }
    __syncthreads();
  }
  if (tid == 0) { mm[n] = smin[0]; mm[NASSET + n] = smax[0]; }
}

__global__ void k_binidx(const float* __restrict__ x, const float* __restrict__ mm,
                         uint8_t* __restrict__ idx8) {
  __shared__ float xt[256 * 33];
  __shared__ float smin[NASSET], sscale[NASSET];
  const int tid = threadIdx.x;
  const int sbase = blockIdx.x * 256;
  const float* src = x + (size_t)sbase * NASSET;

#if HAVE_ASYNC_LDS
  #pragma unroll
  for (int r = 0; r < 8; ++r) {
    int c  = tid + r * 256;
    int s  = c >> 3;
    int a4 = (c & 7) << 2;
    const float* g = src + (size_t)c * 4;
    float* l = &xt[s * 33 + a4];
    __builtin_amdgcn_global_load_async_to_lds_b128(
        (gbl_b128_p)g, (lds_b128_p)l, 0, 0);
  }
#else
  #pragma unroll
  for (int r = 0; r < 32; ++r) {
    int e = tid + r * 256;
    xt[(e >> 5) * 33 + (e & 31)] = src[e];
  }
#endif
  if (tid < NASSET) {
    float mn = mm[tid], mx = mm[NASSET + tid];
    smin[tid] = mn;
    sscale[tid] = (float)NBINS / (mx - mn);
  }
#if HAVE_ASYNC_LDS
  __builtin_amdgcn_s_wait_asynccnt(0);
#endif
  __syncthreads();
  #pragma unroll
  for (int r = 0; r < 8; ++r) {
    int d  = tid + r * 256;
    int a  = d >> 6;
    int s4 = (d & 63) << 2;
    float mn = smin[a], sc = sscale[a];
    uint32_t pack = 0;
    #pragma unroll
    for (int b = 0; b < 4; ++b) {
      float v = xt[(s4 + b) * 33 + a];
      int bin = (int)floorf((v - mn) * sc);
      bin = bin < 0 ? 0 : (bin > NBINS - 1 ? NBINS - 1 : bin);
      pack |= ((uint32_t)bin) << (8 * b);
    }
    *(uint32_t*)(idx8 + (size_t)a * S_TOTAL + sbase + s4) = pack;
  }
}

__global__ void k_onehot(const uint8_t* __restrict__ idx8, uint32_t* __restrict__ A32) {
  const int g   = blockIdx.x * 256 + threadIdx.x;
  const int row = g >> 14;
  const int sd  = g & 16383;
  const int i   = row / NBINS;
  const uint32_t k = (uint32_t)(row - i * NBINS);
  const uint32_t d = ((const uint32_t*)idx8)[(size_t)i * (S_TOTAL / 4) + sd];
  uint32_t out = 0;
  out |= (((d      ) & 255u) == k) ? 0x00000001u : 0u;
  out |= (((d >>  8) & 255u) == k) ? 0x00000100u : 0u;
  out |= (((d >> 16) & 255u) == k) ? 0x00010000u : 0u;
  out |= (((d >> 24)       ) == k) ? 0x01000000u : 0u;
  A32[g] = out;
}

__global__ void k_zero(int* __restrict__ p, int n) {
  int g = blockIdx.x * 256 + threadIdx.x;
  if (g < n) p[g] = 0;
}

__device__ __forceinline__ v8i load_afrag(const uint8_t* p) {
  uint64_t c0 = *(const uint64_t*)(p +  0);
  uint64_t c1 = *(const uint64_t*)(p +  8);
  uint64_t c2 = *(const uint64_t*)(p + 32);
  uint64_t c3 = *(const uint64_t*)(p + 40);
  v8i r;
  r[0] = (int)(uint32_t)c0; r[1] = (int)(uint32_t)(c0 >> 32);
  r[2] = (int)(uint32_t)c1; r[3] = (int)(uint32_t)(c1 >> 32);
  r[4] = (int)(uint32_t)c2; r[5] = (int)(uint32_t)(c2 >> 32);
  r[6] = (int)(uint32_t)c3; r[7] = (int)(uint32_t)(c3 >> 32);
  return r;
}
__device__ __forceinline__ v8i load_bfrag(const uint8_t* p) {
  v4u lo = *(const v4u*)(p +  0);
  v4u hi = *(const v4u*)(p + 32);
  v8i r;
  r[0] = (int)lo[0]; r[1] = (int)lo[1]; r[2] = (int)lo[2]; r[3] = (int)lo[3];
  r[4] = (int)hi[0]; r[5] = (int)hi[1]; r[6] = (int)hi[2]; r[7] = (int)hi[3];
  return r;
}

__global__ __launch_bounds__(256) void k_joint_gemm(const uint8_t* __restrict__ A,
                                                    int* __restrict__ joint) {
  const int lane  = threadIdx.x & 31;
  const int wave  = threadIdx.x >> 5;
  const int l16   = lane & 15;
  const int hi    = lane >> 4;
  const int rowBase = blockIdx.y * 128 + (wave >> 1) * 32;
  const int colBase = blockIdx.x * 128 + (wave &  1) * 64;
  const int k0      = blockIdx.z * KCHUNK;

  const uint8_t* ar0 = A + (size_t)(rowBase + l16) * S_TOTAL + k0 + (hi << 4);
  const uint8_t* ar1 = ar0 + (size_t)16 * S_TOTAL;
  const uint8_t* bc  = A + (size_t)(colBase + l16) * S_TOTAL + k0 + (hi << 4);

  v8i acc[2][4];
  #pragma unroll
  for (int m = 0; m < 2; ++m)
    #pragma unroll
    for (int t = 0; t < 4; ++t) acc[m][t] = (v8i)0;

  #pragma unroll 2
  for (int kk = 0; kk < KCHUNK; kk += 64) {
    v8i a0 = load_afrag(ar0 + kk);
    v8i a1 = load_afrag(ar1 + kk);
    v8i b[4];
    #pragma unroll
    for (int t = 0; t < 4; ++t)
      b[t] = load_bfrag(bc + (size_t)(16 * t) * S_TOTAL + kk);
    #pragma unroll
    for (int t = 0; t < 4; ++t) {
      acc[0][t] = __builtin_amdgcn_wmma_i32_16x16x64_iu8(false, a0, false, b[t], acc[0][t], false, false);
      acc[1][t] = __builtin_amdgcn_wmma_i32_16x16x64_iu8(false, a1, false, b[t], acc[1][t], false, false);
    }
  }

  #pragma unroll
  for (int m = 0; m < 2; ++m) {
    const int r0 = rowBase + 16 * m + (hi << 3);
    #pragma unroll
    for (int t = 0; t < 4; ++t) {
      const int c0 = colBase + 16 * t + l16;
      #pragma unroll
      for (int v = 0; v < 8; ++v)
        atomicAdd(&joint[(size_t)(r0 + v) * ROWS + c0], acc[m][t][v]);
    }
  }
}

__global__ __launch_bounds__(128) void k_vi(const int* __restrict__ joint,
                                            float* __restrict__ out) {
  __shared__ float jp[NBINS * NBINS];
  __shared__ float mx[NBINS], my[NBINS];
  __shared__ float red[128];
  const int i = blockIdx.y, j = blockIdx.x, tid = threadIdx.x;
  const float invS = 1.0f / (float)S_TOTAL;

  for (int e = tid; e < NBINS * NBINS; e += 128) {
    int k = e / NBINS, l = e - k * NBINS;
    jp[e] = (float)joint[(size_t)(i * NBINS + k) * ROWS + (j * NBINS + l)] * invS;
  }
  __syncthreads();
  if (tid < NBINS) {
    float s = 0.f;
    for (int l = 0; l < NBINS; ++l) s += jp[tid * NBINS + l];
    mx[tid] = s;
  } else if (tid >= 32 && tid < 32 + NBINS) {
    int l = tid - 32;
    float s = 0.f;
    for (int k = 0; k < NBINS; ++k) s += jp[k * NBINS + l];
    my[l] = s;
  }
  __syncthreads();
  float part = 0.f;
  for (int e = tid; e < NBINS * NBINS; e += 128) {
    int k = e / NBINS, l = e - k * NBINS;
    float p = jp[e];
    if (p > 0.f) part += p * logf(p / (mx[k] * my[l]) + EPS_F);
  }
  red[tid] = part;
  __syncthreads();
  for (int off = 64; off > 0; off >>= 1) {
    if (tid < off) red[tid] += red[tid + off];
    __syncthreads();
  }
  if (tid == 0) {
    float Hi = 0.f, Hj = 0.f;
    for (int k = 0; k < NBINS; ++k) {
      Hi -= mx[k] * logf(mx[k] + EPS_F);
      Hj -= my[k] * logf(my[k] + EPS_F);
    }
    float vi = (i == j) ? 0.f : (Hi + Hj - 2.f * red[0]);
    for (int b = 0; b < 4; ++b)
      out[(size_t)b * NASSET * NASSET + i * NASSET + j] = vi;
  }
}

extern "C" void kernel_launch(void* const* d_in, const int* in_sizes, int n_in,
                              void* d_out, int out_size, void* d_ws, size_t ws_size,
                              hipStream_t stream) {
  (void)in_sizes; (void)n_in; (void)out_size; (void)ws_size;
  const float* x = (const float*)d_in[0];
  float* out     = (float*)d_out;
  uint8_t* ws    = (uint8_t*)d_ws;

  float*   mm    = (float*)(ws + WS_MINMAX);
  uint8_t* idx8  = ws + WS_IDX8;
  uint8_t* A     = ws + WS_A;
  int*     joint = (int*)(ws + WS_JOINT);

  k_minmax<<<NASSET, 256, 0, stream>>>(x, mm);
  k_binidx<<<S_TOTAL / 256, 256, 0, stream>>>(x, mm, idx8);
  k_onehot<<<(ROWS * (S_TOTAL / 4)) / 256, 256, 0, stream>>>(idx8, (uint32_t*)A);
  k_zero<<<(ROWS * ROWS + 255) / 256, 256, 0, stream>>>(joint, ROWS * ROWS);

  dim3 gg(ROWS / 128, ROWS / 128, SPLITK);
  k_joint_gemm<<<gg, 256, 0, stream>>>(A, joint);

  dim3 gf(NASSET, NASSET, 1);
  k_vi<<<gf, 128, 0, stream>>>(joint, out);
}
